// RNNAgent_31172872634722
// MI455X (gfx1250) — compile-verified, matrix operand order checked by reading, not run on hardware
//
#include <hip/hip_runtime.h>
#include <hip/hip_bf16.h>

typedef __attribute__((ext_vector_type(16))) _Float16 v16h;
typedef __attribute__((ext_vector_type(8)))  float    v8f;

#define T_STEPS   4096
#define HID       64
#define NA        7
#define RIN       14408
#define RIN_PAD   14432
#define CONV_OUT  14400

// ---- workspace layout (bytes) ----
#define FEAT_OFF  ((size_t)0)
#define FEAT_BYTES ((size_t)T_STEPS * RIN_PAD * 2)            // 118,226,944
#define WIH_OFF   (FEAT_OFF + FEAT_BYTES)
#define WIH_BYTES ((size_t)192 * RIN_PAD * 2)                 // 5,541,888
#define GX_OFF    (WIH_OFF + WIH_BYTES)
#define GX_BYTES  ((size_t)T_STEPS * 192 * 4)                 // 3,145,728
#define HS_OFF    (GX_OFF + GX_BYTES)

// ---- LDS layout for conv kernel (halves) ----
#define IMG_H   12288                 // [64][64][3] NHWC fp16
#define C1_H    30752                 // [961][32]  position-major conv1 out
#define W1_H    1024                  // [32][32]   conv1 weights, K pad 27->32
#define W2_H    18432                 // [64][288]  conv2 weights, (ky,kx,ic) order
#define LDS_HALVES (IMG_H + C1_H + W1_H + W2_H)
#define LDS_BYTES  (LDS_HALVES * 2)   // 124,992 B

// ---------------------------------------------------------------------------
// WMMA helpers (layouts per CDNA5 ISA 7.12.2, wave32)
// ---------------------------------------------------------------------------
__device__ __forceinline__ v8f wmma_f16(v16h a, v16h b, v8f c) {
  return __builtin_amdgcn_wmma_f32_16x16x32_f16(
      /*neg_a=*/false, a, /*neg_b=*/false, b,
      /*c_mod=*/(short)0, c, /*reuse_a=*/false, /*reuse_b=*/false);
}

// Fragment from a per-lane row base pointer (kb already folded in by caller):
// elements 0..7 = p[0..7], 8..15 = p[16..23]  -> two 16B contiguous loads.
__device__ __forceinline__ v16h frag_load_ptr(const _Float16* p) {
  v16h f;
#pragma unroll
  for (int e = 0; e < 8; ++e) { f[e] = p[e]; f[8 + e] = p[16 + e]; }
  return f;
}

// Fragment from row-major [M x K], leading dim ld, tile origin (m0,k0).
__device__ __forceinline__ v16h frag_load_rm(const _Float16* base, int ld,
                                             int m0, int k0, int lane) {
  return frag_load_ptr(base + (m0 + (lane & 15)) * ld + k0 + ((lane >> 4) << 3));
}

// ---------------------------------------------------------------------------
// Kernel 1: pack W_ih (192 x 14408 fp32) -> fp16, K padded to 14432
// ---------------------------------------------------------------------------
__global__ __launch_bounds__(256) void cvt_wih(const float* __restrict__ w,
                                               _Float16* __restrict__ out) {
  int i = blockIdx.x * 256 + threadIdx.x;          // exact grid: 192*14432
  int j = i / RIN_PAD;
  int k = i - j * RIN_PAD;
  out[i] = (k < RIN) ? (_Float16)w[j * RIN + k] : (_Float16)0.0f;
}

// ---------------------------------------------------------------------------
// Kernel 2: fused preprocess + conv1 + conv2 (+emb append), one WG per frame.
// All WMMA gathers are branch-free: padded K elements hit zero weights and
// padded N columns are never stored, so clamped in-bounds reads are safe.
// ---------------------------------------------------------------------------
__global__ __launch_bounds__(256) void conv_fused(
    const float* __restrict__ image, const int* __restrict__ dir,
    const float* __restrict__ c1w, const float* __restrict__ c1b,
    const float* __restrict__ c2w, const float* __restrict__ c2b,
    const float* __restrict__ emb, _Float16* __restrict__ feat) {
  extern __shared__ _Float16 smem[];
  _Float16* img_lds = smem;                       // [64][64][3]
  _Float16* c1_pm   = smem + IMG_H;               // [961][32]
  _Float16* w1_lds  = smem + IMG_H + C1_H;        // [32][32]
  _Float16* w2_lds  = smem + IMG_H + C1_H + W1_H; // [64][288]

  const int t    = blockIdx.x;
  const int tid  = threadIdx.x;
  const int lane = tid & 31;
  const int wave = tid >> 5;
  const int kb   = (lane >> 4) << 3;              // 0 or 8

  // ---- stage LDS ----
  const float* img = image + (size_t)t * IMG_H;
  for (int i = tid; i < IMG_H; i += 256)
    img_lds[i] = (_Float16)(img[i] * (1.0f / 255.0f));
  for (int i = tid; i < W1_H; i += 256) {
    int o = i >> 5, k = i & 31;                   // K order: c*9 + ky*3 + kx
    w1_lds[i] = (k < 27) ? (_Float16)c1w[o * 27 + k] : (_Float16)0.0f;
  }
  for (int i = tid; i < W2_H; i += 256) {
    int o = i / 288, k2 = i - o * 288;            // dst K order: kt*32 + ic
    int kt = k2 >> 5, ic = k2 & 31;
    w2_lds[i] = (_Float16)c2w[o * 288 + ic * 9 + kt];
  }
  __syncthreads();

  const v8f vzero = {0.f, 0.f, 0.f, 0.f, 0.f, 0.f, 0.f, 0.f};

  // ---- conv1: M=32 (2 tiles), N=961 (61 tiles), K=32 (27 real) ----
  v16h a1_0 = frag_load_rm(w1_lds, 32, 0, 0, lane);
  v16h a1_1 = frag_load_rm(w1_lds, 32, 16, 0, lane);

  for (int nt = wave; nt < 61; nt += 8) {
    const int col = nt * 16 + (lane & 15);        // output position 0..975
    const int pc  = (col < 961) ? col : 960;      // clamped (branch-free)
    const int oy = pc / 31, ox = pc - oy * 31;
    const int iy = 2 * oy, ix = 2 * ox;
    v16h bf;
#pragma unroll
    for (int e = 0; e < 16; ++e) {
      int k  = kb + e + ((e >= 8) ? 8 : 0);       // ISA K index for element e
      int kc = (k < 27) ? k : 26;                 // pad K -> zero weight anyway
      int c = kc / 9, r = kc - c * 9, ky = r / 3, kx = r - ky * 3;
      bf[e] = img_lds[((iy + ky) * 64 + (ix + kx)) * 3 + c];
    }
    v8f acc0 = wmma_f16(a1_0, bf, vzero);
    v8f acc1 = wmma_f16(a1_1, bf, vzero);
    if (col < 961) {
      const int mb = (lane >> 4) << 3;
      _Float16* row = c1_pm + col * 32;           // position-major
#pragma unroll
      for (int v = 0; v < 8; ++v)
        row[mb + v] = (_Float16)fmaxf(acc0[v] + c1b[mb + v], 0.0f);
#pragma unroll
      for (int v = 0; v < 8; ++v)
        row[16 + mb + v] = (_Float16)fmaxf(acc1[v] + c1b[16 + mb + v], 0.0f);
    }
  }
  __syncthreads();

  // ---- conv2: M=64 (4 tiles), N=225 (15 tiles), K=288 (9 steps) ----
  // K-step kt <-> (ky,kx); B fragment = 2 contiguous b128 LDS loads per lane.
  for (int nt = wave; nt < 15; nt += 8) {
    const int col = nt * 16 + (lane & 15);        // output position 0..239
    const int pc  = (col < 225) ? col : 224;      // clamped (branch-free)
    const int oy = pc / 15, ox = pc - oy * 15;
    const int by = 2 * oy, bx = 2 * ox;
    v8f acc[4] = {vzero, vzero, vzero, vzero};
#pragma unroll
    for (int kt = 0; kt < 9; ++kt) {
      const int ky = kt / 3, kx = kt - ky * 3;    // constants after unroll
      const _Float16* rb = c1_pm + ((by + ky) * 31 + (bx + kx)) * 32 + kb;
      v16h bf = frag_load_ptr(rb);
#pragma unroll
      for (int mi = 0; mi < 4; ++mi) {
        v16h af = frag_load_rm(w2_lds, 288, mi * 16, kt * 32, lane);
        acc[mi] = wmma_f16(af, bf, acc[mi]);
      }
    }
    if (col < 225) {
      const int mb = (lane >> 4) << 3;
      _Float16* frow = feat + (size_t)t * RIN_PAD;
#pragma unroll
      for (int mi = 0; mi < 4; ++mi)
#pragma unroll
        for (int v = 0; v < 8; ++v) {
          int ch = mi * 16 + mb + v;
          frow[ch * 225 + col] = (_Float16)fmaxf(acc[mi][v] + c2b[ch], 0.0f);
        }
    }
  }

  // ---- append direction embedding + zero pad (feat[14400..14431]) ----
  if (tid < 32) {
    int d = dir[t];
    float v = (tid < 8) ? emb[d * 8 + tid] : 0.0f;
    feat[(size_t)t * RIN_PAD + CONV_OUT + tid] = (_Float16)v;
  }
}

// ---------------------------------------------------------------------------
// Kernel 3: gx = feat @ W_ih^T + b_ih   (M=4096, N=192, K=14432)
// WG = 4 waves, one 16-row t-tile per WG, 3 N-tiles per wave, 451 K-steps.
// ---------------------------------------------------------------------------
__global__ __launch_bounds__(128) void gx_gemm(
    const _Float16* __restrict__ feat, const _Float16* __restrict__ wih,
    const float* __restrict__ bih, float* __restrict__ gx) {
  const int lane = threadIdx.x & 31;
  const int wave = threadIdx.x >> 5;               // 0..3
  const int m0   = blockIdx.x * 16;                // t-row base
  const v8f vzero = {0.f, 0.f, 0.f, 0.f, 0.f, 0.f, 0.f, 0.f};
  v8f acc[3] = {vzero, vzero, vzero};

  for (int ks = 0; ks < RIN_PAD / 32; ++ks) {
    const int k0 = ks * 32;
    v16h af = frag_load_rm(feat, RIN_PAD, m0, k0, lane);
#pragma unroll
    for (int q = 0; q < 3; ++q) {
      v16h bf = frag_load_rm(wih, RIN_PAD, (wave * 3 + q) * 16, k0, lane);
      acc[q] = wmma_f16(af, bf, acc[q]);
    }
  }
  const int mb = (lane >> 4) << 3;
#pragma unroll
  for (int q = 0; q < 3; ++q) {
    int j = (wave * 3 + q) * 16 + (lane & 15);     // gate index 0..191
    float b = bih[j];
#pragma unroll
    for (int v = 0; v < 8; ++v) {
      int tr = m0 + mb + v;
      gx[tr * 192 + j] = acc[q][v] + b;
    }
  }
}

// ---------------------------------------------------------------------------
// Kernel 4: sequential GRU scan, one workgroup, 192 threads (one gate row ea.)
// ---------------------------------------------------------------------------
__global__ __launch_bounds__(192) void gru_scan(
    const float* __restrict__ gx, const float* __restrict__ whh,
    const float* __restrict__ bhh, float* __restrict__ hs) {
  __shared__ float h[HID];
  __shared__ float s[192];
  const int j = threadIdx.x;

  float w[HID];
#pragma unroll
  for (int k = 0; k < HID; ++k) w[k] = whh[j * HID + k];
  const float b = bhh[j];
  if (j < HID) h[j] = 0.0f;
  __syncthreads();

  for (int t = 0; t < T_STEPS; ++t) {
    float acc = b;                                  // gh_j = b_hh[j] + W_hh[j,:]·h
#pragma unroll
    for (int k = 0; k < HID; ++k) acc += w[k] * h[k];
    s[j] = acc;
    __syncthreads();
    if (j < HID) {
      const float* g = gx + t * 192;
      float r  = 1.0f / (1.0f + __expf(-(g[j] + s[j])));
      float z  = 1.0f / (1.0f + __expf(-(g[HID + j] + s[HID + j])));
      float n  = tanhf(g[2 * HID + j] + r * s[2 * HID + j]);
      float hn = (1.0f - z) * n + z * h[j];
      hs[t * HID + j] = hn;
      h[j] = hn;
    }
    __syncthreads();
  }
}

// ---------------------------------------------------------------------------
// Kernel 5: actor/critic heads + softmax, one thread per timestep
// ---------------------------------------------------------------------------
__global__ __launch_bounds__(256) void heads(
    const float* __restrict__ hs, const float* __restrict__ aw,
    const float* __restrict__ ab, const float* __restrict__ cw,
    const float* __restrict__ cb, float* __restrict__ out) {
  int t = blockIdx.x * 256 + threadIdx.x;
  if (t >= T_STEPS) return;
  const float* h = hs + t * HID;
  float logit[NA], m = -3.402823466e38f;
#pragma unroll
  for (int i = 0; i < NA; ++i) {
    float a = ab[i];
#pragma unroll
    for (int k = 0; k < HID; ++k) a += aw[i * HID + k] * h[k];
    logit[i] = a;
    m = fmaxf(m, a);
  }
  float sum = 0.0f;
#pragma unroll
  for (int i = 0; i < NA; ++i) { logit[i] = __expf(logit[i] - m); sum += logit[i]; }
  float inv = 1.0f / sum;
#pragma unroll
  for (int i = 0; i < NA; ++i) out[t * NA + i] = logit[i] * inv;
  float v = cb[0];
#pragma unroll
  for (int k = 0; k < HID; ++k) v += cw[k] * h[k];
  out[T_STEPS * NA + t] = v;
}

// ---------------------------------------------------------------------------
extern "C" void kernel_launch(void* const* d_in, const int* in_sizes, int n_in,
                              void* d_out, int out_size, void* d_ws, size_t ws_size,
                              hipStream_t stream) {
  const float* image = (const float*)d_in[0];
  const int*   dir   = (const int*)d_in[1];
  const float* c1w   = (const float*)d_in[2];
  const float* c1b   = (const float*)d_in[3];
  const float* c2w   = (const float*)d_in[4];
  const float* c2b   = (const float*)d_in[5];
  const float* embp  = (const float*)d_in[6];
  const float* wih   = (const float*)d_in[7];
  const float* bih   = (const float*)d_in[8];
  const float* whh   = (const float*)d_in[9];
  const float* bhh   = (const float*)d_in[10];
  const float* aw    = (const float*)d_in[11];
  const float* ab    = (const float*)d_in[12];
  const float* cw    = (const float*)d_in[13];
  const float* cb    = (const float*)d_in[14];

  char* ws = (char*)d_ws;
  _Float16* feat  = (_Float16*)(ws + FEAT_OFF);
  _Float16* wih_h = (_Float16*)(ws + WIH_OFF);
  float*    gx    = (float*)(ws + GX_OFF);
  float*    hsbuf = (float*)(ws + HS_OFF);

  cvt_wih<<<(192 * RIN_PAD) / 256, 256, 0, stream>>>(wih, wih_h);
  conv_fused<<<T_STEPS, 256, LDS_BYTES, stream>>>(image, dir, c1w, c1b, c2w, c2b,
                                                  embp, feat);
  gx_gemm<<<T_STEPS / 16, 128, 0, stream>>>(feat, wih_h, bih, gx);
  gru_scan<<<1, 192, 0, stream>>>(gx, whh, bhh, hsbuf);
  heads<<<T_STEPS / 256, 256, 0, stream>>>(hsbuf, aw, ab, cw, cb, (float*)d_out);
}
